// TemporalTransformer_73942156968638
// MI455X (gfx1250) — matrix operand order repaired and checked by reading, not compiled, not run on hardware
//
#include <hip/hip_runtime.h>

// ---------------- constants ----------------
#define Sdim 512
#define Ddim 512
#define Fdim 2048
#define Ldim 6
#define Bdim 16
#define Hdim 8
#define BS   8192   // Bdim*Sdim

typedef __attribute__((ext_vector_type(16))) __bf16 v16bf;
typedef __attribute__((ext_vector_type(8)))  float  v8f;

union FragB { unsigned u[8]; v16bf v; };

__device__ __forceinline__ unsigned short f2bf(float f) {
    unsigned u = __float_as_uint(f);
    unsigned r = u + 0x7FFFu + ((u >> 16) & 1u);   // round-to-nearest-even
    return (unsigned short)(r >> 16);
}
__device__ __forceinline__ unsigned pack2(float lo, float hi) {
    return (unsigned)f2bf(lo) | ((unsigned)f2bf(hi) << 16);
}
// A-fragment uint index (16-bit A 16x32, ISA 7.12.2)
__device__ __forceinline__ int a_ui(int i, int half) {
    return half * 4 + (i < 4 ? i : i + 4);
}
// B-fragment uint index (16-bit B 32x16)
__device__ __forceinline__ int b_ui(int i, int half) {
    return half * 4 + (i < 4 ? i : i + 4);
}

// ---- 16-lane-row butterflies in VALU via v_permlane16_b32 (no LDS) ----
__device__ __forceinline__ float pl16(float x, int s0, int s1) {
    int v = __float_as_int(x);
    return __int_as_float(
        __builtin_amdgcn_permlane16(v, v, s0, s1, false, false));
}
__device__ __forceinline__ float rmax16(float x) {
    x = fmaxf(x, pl16(x, 0x67452301, (int)0xEFCDAB89));   // xor 1
    x = fmaxf(x, pl16(x, 0x45670123, (int)0xCDEF89AB));   // xor 2
    x = fmaxf(x, pl16(x, 0x01234567, (int)0x89ABCDEF));   // xor 4
    x = fmaxf(x, pl16(x, (int)0xFEDCBA98, 0x76543210));   // xor 8
    return x;
}
__device__ __forceinline__ float rsum16(float x) {
    x += pl16(x, 0x67452301, (int)0xEFCDAB89);
    x += pl16(x, 0x45670123, (int)0xCDEF89AB);
    x += pl16(x, 0x01234567, (int)0x89ABCDEF);
    x += pl16(x, (int)0xFEDCBA98, 0x76543210);
    return x;
}

// -------- f32 -> bf16 weight convert, transposed into 32-row K panels ------
// dst[mtx][k/32][n][k%32]  -> a (k0,n0) tile of width TN is TN*32 contiguous
__global__ void cvtT_kernel(const float* __restrict__ src,
                            unsigned short* __restrict__ dst,
                            int K, int N, int total) {
    int idx = blockIdx.x * 256 + threadIdx.x;
    if (idx >= total) return;
    int per = K * N;
    int mtx = idx / per, r = idx - mtx * per;
    int k = r / N, n = r - k * N;
    dst[(size_t)mtx * per + (size_t)(k >> 5) * N * 32 + n * 32 + (k & 31)] =
        f2bf(src[idx]);
}

// ---------------- relative-position mean table ----------------
__global__ void posmean_kernel(const float* __restrict__ rel_emb,
                               float* __restrict__ pm) {
    int idx = blockIdx.x * 256 + threadIdx.x;   // S*D threads
    int d = idx & (Ddim - 1);
    int j = idx >> 9;
    float cn = (j - 31 > 0) ? (float)(j - 31) : 0.f;          // delta <= -32
    float cp = (480 - j > 0) ? (float)(480 - j) : 0.f;        // delta >= +32
    float acc = cn * rel_emb[d] + cp * rel_emb[64 * Ddim + d];
    for (int delta = -31; delta <= 31; delta++) {
        int i = j + delta;
        if (i >= 0 && i < Sdim) acc += rel_emb[(delta + 32) * Ddim + d];
    }
    pm[idx] = acc * (1.f / (float)Sdim);
}

__global__ void add_pos_kernel(float* __restrict__ h,
                               const float* __restrict__ pm) {
    int idx = blockIdx.x * 256 + threadIdx.x;   // BS*D threads
    int d = idx & (Ddim - 1);
    int m = idx >> 9;
    int s = m & (Sdim - 1);
    h[idx] += pm[s * Ddim + d];
}

// ---------------- generic bf16-WMMA GEMM ----------------
// C[M,N] = act(A[M,K](f32) * W(panels,bf16) + bias [+ residual])
// block tile 128x128, 8 waves (4Mx2N), wave = 32x64 (2x4 WMMA)
// double-buffered LDS; weight tiles arrive via async global->LDS copies
template <bool HAS_RES, bool RELU>
__global__ __launch_bounds__(256) void gemm_bf16(
    const float* __restrict__ A, const unsigned short* __restrict__ Wp,
    const float* __restrict__ bias, const float* __restrict__ res,
    float* __restrict__ C, int M, int N, int K) {
    __shared__ __align__(16) unsigned short As[2][128 * 32];
    __shared__ __align__(16) unsigned short Ws[2][128 * 32];
    const int m0 = blockIdx.x * 128;
    const int n0 = blockIdx.y * 128;
    const int tid = threadIdx.x;
    const int w = tid >> 5, l = tid & 31;
    const int half = l >> 4, lm = l & 15;
    const int waveM = (w & 3) * 32, waveN = (w >> 2) * 64;
    const int nk = K >> 5;
    v8f acc[2][4] = {};

    // A tile: 128x32 f32 -> bf16, packed pairs, 2x b128 LDS stores
    auto loadA = [&](int kt, int buf) {
        int row = tid >> 1, cc = (tid & 1) * 16;
        const float* src = A + (size_t)(m0 + row) * K + kt * 32 + cc;
        uint4 w0, w1;
        w0.x = pack2(src[0], src[1]);   w0.y = pack2(src[2], src[3]);
        w0.z = pack2(src[4], src[5]);   w0.w = pack2(src[6], src[7]);
        w1.x = pack2(src[8], src[9]);   w1.y = pack2(src[10], src[11]);
        w1.z = pack2(src[12], src[13]); w1.w = pack2(src[14], src[15]);
        uint4* dst = (uint4*)&As[buf][row * 32 + cc];
        dst[0] = w0; dst[1] = w1;
    };
    // W tile: one contiguous 8KB panel slice -> async copy to LDS
    auto loadW = [&](int kt, int buf) {
        const unsigned short* gp = Wp + (size_t)kt * N * 32 + (size_t)n0 * 32;
        unsigned ldsb = (unsigned)(size_t)(&Ws[buf][0]) + tid * 16;
        unsigned voff = (unsigned)(tid * 16);
        asm volatile("global_load_async_to_lds_b128 %0, %1, %2 offset:0"
                     :: "v"(ldsb), "v"(voff), "s"(gp) : "memory");
        asm volatile("global_load_async_to_lds_b128 %0, %1, %2 offset:4096"
                     :: "v"(ldsb), "v"(voff), "s"(gp) : "memory");
    };

    loadA(0, 0);
    loadW(0, 0);
    asm volatile("s_wait_asynccnt 0" ::: "memory");
    __syncthreads();

    for (int kt = 0; kt < nk; kt++) {
        int cur = kt & 1, nxt = cur ^ 1;
        if (kt + 1 < nk) { loadA(kt + 1, nxt); loadW(kt + 1, nxt); }

        const unsigned* Au = (const unsigned*)As[cur];
        const unsigned* Wu = (const unsigned*)Ws[cur];
        FragB a[2], b[4];
#pragma unroll
        for (int mi = 0; mi < 2; mi++) {
            int row = waveM + mi * 16 + lm;
#pragma unroll
            for (int i = 0; i < 8; i++) a[mi].u[i] = Au[row * 16 + a_ui(i, half)];
        }
#pragma unroll
        for (int ni = 0; ni < 4; ni++) {
            int col = waveN + ni * 16 + lm;
#pragma unroll
            for (int i = 0; i < 8; i++) b[ni].u[i] = Wu[col * 16 + b_ui(i, half)];
        }
#pragma unroll
        for (int mi = 0; mi < 2; mi++)
#pragma unroll
            for (int ni = 0; ni < 4; ni++)
                acc[mi][ni] = __builtin_amdgcn_wmma_f32_16x16x32_bf16(
                    false, a[mi].v, false, b[ni].v, (short)0, acc[mi][ni],
                    false, false);

        asm volatile("s_wait_asynccnt 0" ::: "memory");
        __syncthreads();
    }

#pragma unroll
    for (int mi = 0; mi < 2; mi++)
#pragma unroll
        for (int ni = 0; ni < 4; ni++) {
            int n = n0 + waveN + ni * 16 + lm;
            float bv = bias[n];
#pragma unroll
            for (int r = 0; r < 8; r++) {
                int m = m0 + waveM + mi * 16 + half * 8 + r;
                float v = acc[mi][ni][r] + bv;
                if (HAS_RES) v += res[(size_t)m * N + n];
                if (RELU) v = v > 0.f ? v : 0.f;
                C[(size_t)m * N + n] = v;
            }
        }
}

// ---------------- fused flash attention ----------------
// Q,K,V stored as (B*S, D) f32; head hh occupies cols [hh*64, hh*64+64)
// grid: (B*H, S/128); block 256; wave handles 16 query rows
__global__ __launch_bounds__(256) void attn_kernel(
    const float* __restrict__ Q, const float* __restrict__ Kb,
    const float* __restrict__ Vb, float* __restrict__ O) {
    __shared__ __align__(16) unsigned short ks[32 * 64];    // [key][dk]
    __shared__ __align__(16) unsigned short vs[64 * 32];    // [dk][key-pairs]
    __shared__ __align__(16) unsigned short ps[8][16 * 32]; // per-wave P staging
    const int bh = blockIdx.x, qt = blockIdx.y;
    const int b = bh >> 3, hh = bh & 7;
    const int tid = threadIdx.x, w = tid >> 5, l = tid & 31;
    const int half = l >> 4, lm = l & 15;
    const size_t base = (size_t)b * Sdim * Ddim + hh * 64;

    // preload Q fragments (dk split into two 32-chunks)
    FragB qa[2];
    {
        int qrow = qt * 128 + w * 16 + lm;
        const float* qp = Q + base + (size_t)qrow * Ddim;
#pragma unroll
        for (int f = 0; f < 2; f++)
#pragma unroll
            for (int i = 0; i < 8; i++) {
                int p = a_ui(i, half) * 2;
                qa[f].u[i] = pack2(qp[f * 32 + p], qp[f * 32 + p + 1]);
            }
    }
    float mrow[8], lrow[8];
#pragma unroll
    for (int r = 0; r < 8; r++) { mrow[r] = -1e30f; lrow[r] = 0.f; }
    v8f o[4] = {};
    const unsigned* ksu = (const unsigned*)ks;
    const unsigned* vsu = (const unsigned*)vs;
    unsigned* vsw = (unsigned*)vs;
    const unsigned* psu = (const unsigned*)ps[w];

    for (int kt = 0; kt < Sdim / 32; kt++) {
        __syncthreads();
        {   // K tile [key][dk]: 4 packed bf16 pairs -> one b128 store
            int key = tid >> 3, dc = (tid & 7) * 8;
            const float* srcK = Kb + base + (size_t)(kt * 32 + key) * Ddim + dc;
            uint4 kkv;
            kkv.x = pack2(srcK[0], srcK[1]); kkv.y = pack2(srcK[2], srcK[3]);
            kkv.z = pack2(srcK[4], srcK[5]); kkv.w = pack2(srcK[6], srcK[7]);
            *(uint4*)&ks[key * 64 + dc] = kkv;
        }
        {   // V tile transposed [dk][key]: pack key pairs per dword
            int kp = tid & 15, dc = (tid >> 4) * 4;
            const float* v0 = Vb + base + (size_t)(kt * 32 + 2 * kp) * Ddim + dc;
            const float* v1 = v0 + Ddim;
#pragma unroll
            for (int j = 0; j < 4; j++)
                vsw[(dc + j) * 16 + kp] = pack2(v0[j], v1[j]);
        }
        __syncthreads();

        // scores: 16 q-rows x 32 keys
        v8f sc[2] = {};
#pragma unroll
        for (int ni = 0; ni < 2; ni++) {
            int key = ni * 16 + lm;
#pragma unroll
            for (int f = 0; f < 2; f++) {
                FragB bb;
#pragma unroll
                for (int i = 0; i < 8; i++)
                    bb.u[i] = ksu[key * 32 + f * 16 + b_ui(i, half)];
                sc[ni] = __builtin_amdgcn_wmma_f32_16x16x32_bf16(
                    false, qa[f].v, false, bb.v, (short)0, sc[ni], false, false);
            }
        }

        // online softmax (row stats via permlane16 butterflies, pure VALU)
        float p0[8], p1[8];
#pragma unroll
        for (int r = 0; r < 8; r++) {
            float v0 = sc[0][r] * 0.125f, v1 = sc[1][r] * 0.125f; // 1/sqrt(64)
            float rmax = rmax16(fmaxf(v0, v1));
            float mnew = fmaxf(mrow[r], rmax);
            float corr = __expf(mrow[r] - mnew);
            float e0 = __expf(v0 - mnew), e1 = __expf(v1 - mnew);
            float rs = rsum16(e0 + e1);
            lrow[r] = lrow[r] * corr + rs;
            mrow[r] = mnew;
#pragma unroll
            for (int ni = 0; ni < 4; ni++) o[ni][r] *= corr;
            p0[r] = e0; p1[r] = e1;
        }

        // redistribute P (C layout -> A layout) through per-wave LDS
#pragma unroll
        for (int r = 0; r < 8; r++) {
            int m = half * 8 + r;
            ps[w][m * 32 + lm]      = f2bf(p0[r]);
            ps[w][m * 32 + 16 + lm] = f2bf(p1[r]);
        }
        asm volatile("s_wait_dscnt 0" ::: "memory");
        FragB pa;
#pragma unroll
        for (int i = 0; i < 8; i++) pa.u[i] = psu[lm * 16 + a_ui(i, half)];

#pragma unroll
        for (int ni = 0; ni < 4; ni++) {
            FragB vv;
            int dk = ni * 16 + lm;
#pragma unroll
            for (int i = 0; i < 8; i++) vv.u[i] = vsu[dk * 16 + b_ui(i, half)];
            o[ni] = __builtin_amdgcn_wmma_f32_16x16x32_bf16(
                false, pa.v, false, vv.v, (short)0, o[ni], false, false);
        }
    }

    // normalize and store context
#pragma unroll
    for (int ni = 0; ni < 4; ni++)
#pragma unroll
        for (int r = 0; r < 8; r++) {
            int s = qt * 128 + w * 16 + half * 8 + r;
            O[base + (size_t)s * Ddim + ni * 16 + lm] = o[ni][r] / lrow[r];
        }
}

// ---------------- layernorm: one wave per row (D=512) ----------------
__global__ __launch_bounds__(256) void ln_kernel(
    const float* __restrict__ x, const float* __restrict__ g,
    const float* __restrict__ bb, float* __restrict__ y) {
    int row = blockIdx.x * 8 + (threadIdx.x >> 5);
    int lane = threadIdx.x & 31;
    const float* xr = x + (size_t)row * Ddim;
    float s = 0.f, s2 = 0.f, vals[16];
#pragma unroll
    for (int i = 0; i < 16; i++) {
        float v = xr[lane + i * 32];
        vals[i] = v; s += v; s2 += v * v;
    }
#pragma unroll
    for (int off = 16; off; off >>= 1) {
        s  += __shfl_xor(s, off, 32);
        s2 += __shfl_xor(s2, off, 32);
    }
    float mu = s * (1.f / Ddim);
    float var = s2 * (1.f / Ddim) - mu * mu;
    float inv = rsqrtf(var + 1e-5f);
#pragma unroll
    for (int i = 0; i < 16; i++) {
        int d = lane + i * 32;
        y[(size_t)row * Ddim + d] = (vals[i] - mu) * inv * g[d] + bb[d];
    }
}

// ---------------- final projection D -> 1 ----------------
__global__ __launch_bounds__(256) void out_kernel(
    const float* __restrict__ h, const float* __restrict__ ow,
    const float* __restrict__ ob, float* __restrict__ out) {
    int row = blockIdx.x * 8 + (threadIdx.x >> 5);
    int lane = threadIdx.x & 31;
    float s = 0.f;
#pragma unroll
    for (int i = 0; i < 16; i++) {
        int d = lane + i * 32;
        s += h[(size_t)row * Ddim + d] * ow[d];
    }
#pragma unroll
    for (int off = 16; off; off >>= 1) s += __shfl_xor(s, off, 32);
    if (lane == 0) out[row] = s + ob[0];
}

// ---------------- host orchestration ----------------
extern "C" void kernel_launch(void* const* d_in, const int* in_sizes, int n_in,
                              void* d_out, int out_size, void* d_ws, size_t ws_size,
                              hipStream_t stream) {
    (void)in_sizes; (void)n_in; (void)out_size; (void)ws_size;
    const float* x       = (const float*)d_in[0];
    const float* in_W    = (const float*)d_in[1];
    const float* in_b    = (const float*)d_in[2];
    const float* rel_emb = (const float*)d_in[3];
    const float* qkvo_W  = (const float*)d_in[4];
    const float* qkvo_b  = (const float*)d_in[5];
    const float* ln_g    = (const float*)d_in[6];
    const float* ln_b    = (const float*)d_in[7];
    const float* ff_W1   = (const float*)d_in[8];
    const float* ff_b1   = (const float*)d_in[9];
    const float* ff_W2   = (const float*)d_in[10];
    const float* ff_b2   = (const float*)d_in[11];
    const float* out_W   = (const float*)d_in[12];
    const float* out_b   = (const float*)d_in[13];
    float* out = (float*)d_out;

    char* ws = (char*)d_ws;
    size_t off = 0;
    float* pos_mean = (float*)(ws + off);          off += (size_t)Sdim * Ddim * 4;
    unsigned short* in_Wb   = (unsigned short*)(ws + off); off += (size_t)64 * Ddim * 2;
    unsigned short* qkvo_Wb = (unsigned short*)(ws + off); off += (size_t)Ldim * 4 * Ddim * Ddim * 2;
    unsigned short* ff1b    = (unsigned short*)(ws + off); off += (size_t)Ldim * Ddim * Fdim * 2;
    unsigned short* ff2b    = (unsigned short*)(ws + off); off += (size_t)Ldim * Fdim * Ddim * 2;
    off = (off + 255) & ~(size_t)255;
    float* h  = (float*)(ws + off); off += (size_t)BS * Ddim * 4;
    float* t0 = (float*)(ws + off); off += (size_t)BS * Ddim * 4;
    float* t1 = (float*)(ws + off); off += (size_t)BS * Ddim * 4;
    float* t2 = (float*)(ws + off); off += (size_t)BS * Ddim * 4;
    float* t3 = (float*)(ws + off); off += (size_t)BS * Ddim * 4;
    float* fb = (float*)(ws + off); off += (size_t)BS * Fdim * 4;

    // weight conversion into transposed bf16 K-panels (deterministic each call)
    int nInW  = 64 * Ddim;
    int nQKVO = Ldim * 4 * Ddim * Ddim;
    int nFF   = Ldim * Ddim * Fdim;
    cvtT_kernel<<<(nInW + 255) / 256, 256, 0, stream>>>(in_W, in_Wb, 64, Ddim, nInW);
    cvtT_kernel<<<(nQKVO + 255) / 256, 256, 0, stream>>>(qkvo_W, qkvo_Wb, Ddim, Ddim, nQKVO);
    cvtT_kernel<<<(nFF + 255) / 256, 256, 0, stream>>>(ff_W1, ff1b, Ddim, Fdim, nFF);
    cvtT_kernel<<<(nFF + 255) / 256, 256, 0, stream>>>(ff_W2, ff2b, Fdim, Ddim, nFF);
    posmean_kernel<<<(Sdim * Ddim) / 256, 256, 0, stream>>>(rel_emb, pos_mean);

    // input projection + positional add
    gemm_bf16<false, false><<<dim3(BS / 128, Ddim / 128), 256, 0, stream>>>(
        x, in_Wb, in_b, nullptr, h, BS, Ddim, 64);
    add_pos_kernel<<<((size_t)BS * Ddim) / 256, 256, 0, stream>>>(h, pos_mean);

    for (int l = 0; l < Ldim; l++) {
        const unsigned short* Wq = qkvo_Wb + ((size_t)l * 4 + 0) * Ddim * Ddim;
        const unsigned short* Wk = qkvo_Wb + ((size_t)l * 4 + 1) * Ddim * Ddim;
        const unsigned short* Wv = qkvo_Wb + ((size_t)l * 4 + 2) * Ddim * Ddim;
        const unsigned short* Wo = qkvo_Wb + ((size_t)l * 4 + 3) * Ddim * Ddim;
        const float* bq = qkvo_b + ((size_t)l * 4 + 0) * Ddim;
        const float* bk = qkvo_b + ((size_t)l * 4 + 1) * Ddim;
        const float* bv = qkvo_b + ((size_t)l * 4 + 2) * Ddim;
        const float* bo = qkvo_b + ((size_t)l * 4 + 3) * Ddim;

        dim3 gD(BS / 128, Ddim / 128);
        gemm_bf16<false, false><<<gD, 256, 0, stream>>>(h, Wq, bq, nullptr, t0, BS, Ddim, Ddim);
        gemm_bf16<false, false><<<gD, 256, 0, stream>>>(h, Wk, bk, nullptr, t1, BS, Ddim, Ddim);
        gemm_bf16<false, false><<<gD, 256, 0, stream>>>(h, Wv, bv, nullptr, t2, BS, Ddim, Ddim);

        attn_kernel<<<dim3(Bdim * Hdim, Sdim / 128), 256, 0, stream>>>(t0, t1, t2, t3);

        // output projection + residual, then LN1 -> h
        gemm_bf16<true, false><<<gD, 256, 0, stream>>>(t3, Wo, bo, h, t0, BS, Ddim, Ddim);
        ln_kernel<<<BS / 8, 256, 0, stream>>>(
            t0, ln_g + ((size_t)l * 2 + 0) * Ddim, ln_b + ((size_t)l * 2 + 0) * Ddim, h);

        // FFN
        gemm_bf16<false, true><<<dim3(BS / 128, Fdim / 128), 256, 0, stream>>>(
            h, ff1b + (size_t)l * Ddim * Fdim, ff_b1 + (size_t)l * Fdim,
            nullptr, fb, BS, Fdim, Ddim);
        gemm_bf16<true, false><<<gD, 256, 0, stream>>>(
            fb, ff2b + (size_t)l * Fdim * Ddim, ff_b2 + (size_t)l * Ddim,
            h, t0, BS, Ddim, Fdim);
        ln_kernel<<<BS / 8, 256, 0, stream>>>(
            t0, ln_g + ((size_t)l * 2 + 1) * Ddim, ln_b + ((size_t)l * 2 + 1) * Ddim, h);
    }

    out_kernel<<<BS / 8, 256, 0, stream>>>(h, out_W, out_b, out);
}
